// BasisCustBiLSTM_63136019251555
// MI455X (gfx1250) — matrix operand order repaired and checked by reading, not compiled, not run on hardware
//
#include <hip/hip_runtime.h>
#include <hip/hip_bf16.h>
#include <stdint.h>

// Problem dims (compile-time constants from the reference)
#define B_   64
#define T_   256
#define D_   300
#define DP_  320     // D padded to a multiple of 32 for WMMA K
#define H_   256
#define G_   1024    // 4*H
#define N_   8
#define MEMB 64      // embedding dim M
#define KQ_  64

typedef __attribute__((ext_vector_type(16))) __bf16 v16bf;
typedef __attribute__((ext_vector_type(8)))  float  v8f;

union ABFrag { v16bf v; uint4 u[2]; };

// ---------------- workspace layout (bytes, 256-aligned) ----------------
#define OFF_C      ((size_t)0)                                   // 64*8*4          = 2 KB
#define OFF_BIASM  ((size_t)2048)                                // 2*64*1024*4     = 512 KB
#define OFF_XPAD   (OFF_BIASM + (size_t)2*B_*G_*4)               // 64*256*320*2    = 10.5 MB
#define OFF_WIH    (OFF_XPAD + (size_t)B_*T_*DP_*2)              // 2*64*1024*320*2 = 84 MB
#define OFF_WHH    (OFF_WIH  + (size_t)2*B_*G_*DP_*2)            // 2*64*256*1024*2 = 67 MB
#define OFF_XG     (OFF_WHH  + (size_t)2*B_*H_*G_*2)             // 2*64*256*1024*4 = 134 MB

__device__ __forceinline__ float sigmoidf_(float x) {
    return 1.0f / (1.0f + __expf(-x));
}

// ---------------- K1: mixture coefficients c[b][n] ----------------
__global__ void coef_kernel(const int* __restrict__ uidx, const int* __restrict__ iidx,
                            const float* __restrict__ uemb, const float* __restrict__ iemb,
                            const float* __restrict__ W1, const float* __restrict__ b1,
                            const float* __restrict__ W2, float* __restrict__ cws) {
    int b = threadIdx.x;
    if (b >= B_) return;
    const float* ue = uemb + (size_t)uidx[b] * MEMB;
    const float* ie = iemb + (size_t)iidx[b] * MEMB;
    float hbuf[KQ_];
    for (int j = 0; j < KQ_; ++j) {
        float s = b1[j];
        for (int i = 0; i < MEMB; ++i) s = fmaf(ue[i], W1[(size_t)i * KQ_ + j], s);
        for (int i = 0; i < MEMB; ++i) s = fmaf(ie[i], W1[(size_t)(MEMB + i) * KQ_ + j], s);
        hbuf[j] = tanhf(s);
    }
    float lg[N_];
    float mx = -1e30f;
    for (int n = 0; n < N_; ++n) {
        float s = 0.f;
        for (int j = 0; j < KQ_; ++j) s = fmaf(hbuf[j], W2[(size_t)j * N_ + n], s);
        lg[n] = s;
        mx = fmaxf(mx, s);
    }
    float den = 0.f;
    for (int n = 0; n < N_; ++n) { lg[n] = __expf(lg[n] - mx); den += lg[n]; }
    float inv = 1.0f / den;
    for (int n = 0; n < N_; ++n) cws[(size_t)b * N_ + n] = lg[n] * inv;
}

// ---------------- K2a: x -> bf16, K padded to 320 ----------------
__global__ void xpad_kernel(const float* __restrict__ x, __hip_bfloat16* __restrict__ xpad) {
    size_t total = (size_t)B_ * T_ * DP_;
    for (size_t e = (size_t)blockIdx.x * blockDim.x + threadIdx.x; e < total;
         e += (size_t)gridDim.x * blockDim.x) {
        int k = (int)(e % DP_);
        size_t bt = e / DP_;
        float v = (k < D_) ? x[bt * D_ + k] : 0.0f;
        xpad[e] = __float2bfloat16(v);
    }
}

// ---------------- K2b: mix input weights -> bf16 row-major [dir][b][g][320] ----------------
__global__ void mix_ih_kernel(const float* __restrict__ c,
                              const float* __restrict__ Wih, const float* __restrict__ Wih_r,
                              __hip_bfloat16* __restrict__ out) {
    const size_t per = (size_t)B_ * G_ * DP_;
    const size_t total = 2 * per;
    for (size_t e = (size_t)blockIdx.x * blockDim.x + threadIdx.x; e < total;
         e += (size_t)gridDim.x * blockDim.x) {
        int dir = (int)(e / per);
        size_t rem = e % per;
        int b = (int)(rem / ((size_t)G_ * DP_));
        size_t r2 = rem % ((size_t)G_ * DP_);
        int g = (int)(r2 / DP_);
        int k = (int)(r2 % DP_);
        float v = 0.0f;
        if (k < D_) {
            const float* src = dir ? Wih_r : Wih;
            const float* cb = c + (size_t)b * N_;
            #pragma unroll
            for (int n = 0; n < N_; ++n)
                v = fmaf(cb[n], src[(size_t)n * G_ * D_ + (size_t)g * D_ + k], v);
        }
        out[e] = __float2bfloat16(v);
    }
}

// ---- K2c: mix recurrent weights -> bf16 gate-interleaved [dir][b][k][j][q], g = q*256+j ----
__global__ void mix_hh_kernel(const float* __restrict__ c,
                              const float* __restrict__ Whh, const float* __restrict__ Whh_r,
                              __hip_bfloat16* __restrict__ out) {
    const size_t per = (size_t)B_ * H_ * G_;
    const size_t total = 2 * per;
    for (size_t e = (size_t)blockIdx.x * blockDim.x + threadIdx.x; e < total;
         e += (size_t)gridDim.x * blockDim.x) {
        int dir = (int)(e / per);
        size_t rem = e % per;
        int b = (int)(rem / ((size_t)H_ * G_));
        size_t r2 = rem % ((size_t)H_ * G_);
        int k = (int)(r2 / G_);
        int jq = (int)(r2 % G_);
        int j = jq >> 2, q = jq & 3;
        int g = q * H_ + j;
        const float* src = dir ? Whh_r : Whh;
        const float* cb = c + (size_t)b * N_;
        float v = 0.0f;
        #pragma unroll
        for (int n = 0; n < N_; ++n)
            v = fmaf(cb[n], src[(size_t)n * G_ * H_ + (size_t)g * H_ + k], v);
        out[e] = __float2bfloat16(v);
    }
}

// ---------------- K2d: mix bias (fp32) [dir][b][g] ----------------
__global__ void mix_bias_kernel(const float* __restrict__ c,
                                const float* __restrict__ bias, const float* __restrict__ bias_r,
                                float* __restrict__ out) {
    const size_t per = (size_t)B_ * G_;
    const size_t total = 2 * per;
    for (size_t e = (size_t)blockIdx.x * blockDim.x + threadIdx.x; e < total;
         e += (size_t)gridDim.x * blockDim.x) {
        int dir = (int)(e / per);
        size_t rem = e % per;
        int b = (int)(rem / G_);
        int g = (int)(rem % G_);
        const float* src = dir ? bias_r : bias;
        const float* cb = c + (size_t)b * N_;
        float v = 0.0f;
        #pragma unroll
        for (int n = 0; n < N_; ++n) v = fmaf(cb[n], src[(size_t)n * G_ + g], v);
        out[e] = v;
    }
}

// ---------------- K3: input projection via WMMA bf16, software-pipelined ----------------
// Xg[dir][b][t][g] = bias_mix + sum_k xpad[b][t][k] * Wih_mix[dir][b][g][k]
// One wave per 32(t) x 64(g) tile: 2 t-tiles x 4 g-subtiles = 8 accumulators.
// Fragments double-buffered so K-step k+1 loads issue before k's WMMAs consume data.
__global__ __launch_bounds__(32) void input_gemm_kernel(const __hip_bfloat16* __restrict__ xpad,
                                                        const __hip_bfloat16* __restrict__ wih,
                                                        const float* __restrict__ biasm,
                                                        float* __restrict__ Xg) {
    const int lane = threadIdx.x;
    const int col = lane & 15;
    const int hi  = lane >> 4;
    const int z  = blockIdx.z;          // dir*64 + b
    const int b  = z & 63;
    const int t0 = blockIdx.y * 32;     // two 16-row tiles
    const int g0 = blockIdx.x * 64;

    const __hip_bfloat16* xrow0 = xpad + ((size_t)b * T_ + t0 + col) * DP_;
    const __hip_bfloat16* xrow1 = xrow0 + (size_t)16 * DP_;
    const __hip_bfloat16* wbase = wih  + ((size_t)z * G_ + g0 + col) * DP_;

    ABFrag Abuf[2][2];   // [parity][t-tile]
    ABFrag Bbuf[2][4];   // [parity][g-subtile]
    v8f acc[8] = {};     // [m*4 + s]

    // prime K-step 0
    Abuf[0][0].u[0] = *(const uint4*)(xrow0 + hi * 8);
    Abuf[0][0].u[1] = *(const uint4*)(xrow0 + 16 + hi * 8);
    Abuf[0][1].u[0] = *(const uint4*)(xrow1 + hi * 8);
    Abuf[0][1].u[1] = *(const uint4*)(xrow1 + 16 + hi * 8);
    #pragma unroll
    for (int s = 0; s < 4; ++s) {
        const __hip_bfloat16* wrow = wbase + (size_t)s * 16 * DP_;
        Bbuf[0][s].u[0] = *(const uint4*)(wrow + hi * 8);
        Bbuf[0][s].u[1] = *(const uint4*)(wrow + 16 + hi * 8);
    }

    #pragma unroll
    for (int kk = 0; kk < DP_ / 32; ++kk) {
        const int p  = kk & 1;
        const int pn = p ^ 1;
        // issue next K-step's fragment loads before consuming current ones
        if (kk + 1 < DP_ / 32) {
            const int k1 = (kk + 1) * 32;
            Abuf[pn][0].u[0] = *(const uint4*)(xrow0 + k1 + hi * 8);
            Abuf[pn][0].u[1] = *(const uint4*)(xrow0 + k1 + 16 + hi * 8);
            Abuf[pn][1].u[0] = *(const uint4*)(xrow1 + k1 + hi * 8);
            Abuf[pn][1].u[1] = *(const uint4*)(xrow1 + k1 + 16 + hi * 8);
            #pragma unroll
            for (int s = 0; s < 4; ++s) {
                const __hip_bfloat16* wrow = wbase + (size_t)s * 16 * DP_;
                Bbuf[pn][s].u[0] = *(const uint4*)(wrow + k1 + hi * 8);
                Bbuf[pn][s].u[1] = *(const uint4*)(wrow + k1 + 16 + hi * 8);
            }
        }
        // 8 WMMAs on current fragments (B reused across both t-tiles)
        #pragma unroll
        for (int s = 0; s < 4; ++s) {
            acc[s] = __builtin_amdgcn_wmma_f32_16x16x32_bf16(
                false, Abuf[p][0].v, false, Bbuf[p][s].v, (short)0, acc[s], false, false);
            acc[4 + s] = __builtin_amdgcn_wmma_f32_16x16x32_bf16(
                false, Abuf[p][1].v, false, Bbuf[p][s].v, (short)0, acc[4 + s], false, false);
        }
    }

    #pragma unroll
    for (int s = 0; s < 4; ++s) {
        const int gc = g0 + s * 16 + col;
        const float bv = biasm[(size_t)z * G_ + gc];
        #pragma unroll
        for (int m = 0; m < 2; ++m) {
            float* orow = Xg + ((size_t)z * T_ + t0 + m * 16) * G_ + gc;
            #pragma unroll
            for (int r = 0; r < 8; ++r) {
                const int row = r + hi * 8;   // C layout: VGPR r -> M=r (lanes 0-15), M=r+8 (16-31)
                orow[(size_t)row * G_] = acc[m * 4 + s][r] + bv;
            }
        }
    }
}

// ---------------- K4: sequential LSTM recurrence, one WG per (b, dir) ----------------
__global__ __launch_bounds__(256) void lstm_rec_kernel(const float* __restrict__ Xg,
                                                       const __hip_bfloat16* __restrict__ wrec_raw,
                                                       const int* __restrict__ length,
                                                       float* __restrict__ out) {
    const int b = blockIdx.x;
    const int dir = blockIdx.y;
    const int z = dir * B_ + b;
    const int j = threadIdx.x;            // h index this thread owns
    const int len = length[b];

    // weights packed as [k][j][q] bf16, q = gate (i,f,g,o); 4 bf16 = one uint2 per (k,j)
    const uint2* wz = reinterpret_cast<const uint2*>(wrec_raw) + (size_t)z * (H_ * G_ / 4);
    const float* xg = Xg + (size_t)z * T_ * G_;

    __shared__ float hsh[H_];
    float cst = 0.0f;
    hsh[j] = 0.0f;
    __syncthreads();

    for (int s = 0; s < T_; ++s) {
        const int t = dir ? (T_ - 1 - s) : s;
        const float* grow = xg + (size_t)t * G_;
        float a0 = grow[j];                // i gate preact (input part + bias)
        float a1 = grow[H_ + j];           // f
        float a2 = grow[2 * H_ + j];       // g
        float a3 = grow[3 * H_ + j];       // o

        // prefetch next step's gate row (gfx1250 global_prefetch_b8)
        if (s + 1 < T_) {
            const int tn = dir ? (t - 1) : (t + 1);
            __builtin_prefetch(xg + (size_t)tn * G_ + j, 0, 0);
        }

        #pragma unroll 4
        for (int k = 0; k < H_; ++k) {
            const float hk = hsh[k];
            const uint2 w = wz[(size_t)k * (G_ / 4) + j];
            a0 = fmaf(__uint_as_float(w.x << 16),          hk, a0);
            a1 = fmaf(__uint_as_float(w.x & 0xffff0000u),  hk, a1);
            a2 = fmaf(__uint_as_float(w.y << 16),          hk, a2);
            a3 = fmaf(__uint_as_float(w.y & 0xffff0000u),  hk, a3);
        }

        float cy = sigmoidf_(a1) * cst + sigmoidf_(a0) * tanhf(a2);
        float hy = sigmoidf_(a3) * tanhf(cy);
        if (t >= len) { cy = 0.0f; hy = 0.0f; }
        cst = cy;

        out[((size_t)b * T_ + t) * (2 * H_) + (size_t)dir * H_ + j] = hy;

        __syncthreads();
        hsh[j] = hy;
        __syncthreads();
    }
}

// ---------------- host launcher ----------------
extern "C" void kernel_launch(void* const* d_in, const int* in_sizes, int n_in,
                              void* d_out, int out_size, void* d_ws, size_t ws_size,
                              hipStream_t stream) {
    const float* x      = (const float*)d_in[0];
    const int*   length = (const int*)  d_in[1];
    const int*   uidx   = (const int*)  d_in[2];
    const int*   iidx   = (const int*)  d_in[3];
    const float* uemb   = (const float*)d_in[4];
    const float* iemb   = (const float*)d_in[5];
    const float* W1     = (const float*)d_in[6];
    const float* b1     = (const float*)d_in[7];
    const float* W2     = (const float*)d_in[8];
    const float* Wih    = (const float*)d_in[9];
    const float* Whh    = (const float*)d_in[10];
    const float* bias   = (const float*)d_in[11];
    const float* Wih_r  = (const float*)d_in[12];
    const float* Whh_r  = (const float*)d_in[13];
    const float* bias_r = (const float*)d_in[14];
    float* out = (float*)d_out;

    char* ws = (char*)d_ws;
    float*          c_ws   = (float*)         (ws + OFF_C);
    float*          biasm  = (float*)         (ws + OFF_BIASM);
    __hip_bfloat16* xpad   = (__hip_bfloat16*)(ws + OFF_XPAD);
    __hip_bfloat16* wihm   = (__hip_bfloat16*)(ws + OFF_WIH);
    __hip_bfloat16* whhrec = (__hip_bfloat16*)(ws + OFF_WHH);
    float*          Xg     = (float*)         (ws + OFF_XG);

    coef_kernel<<<1, 64, 0, stream>>>(uidx, iidx, uemb, iemb, W1, b1, W2, c_ws);
    xpad_kernel<<<1024, 256, 0, stream>>>(x, xpad);
    mix_bias_kernel<<<256, 256, 0, stream>>>(c_ws, bias, bias_r, biasm);
    mix_ih_kernel<<<8192, 256, 0, stream>>>(c_ws, Wih, Wih_r, wihm);
    mix_hh_kernel<<<8192, 256, 0, stream>>>(c_ws, Whh, Whh_r, whhrec);

    dim3 g3(G_ / 64, T_ / 32, 2 * B_);   // (16, 8, 128)
    input_gemm_kernel<<<g3, 32, 0, stream>>>(xpad, wihm, biasm, Xg);

    dim3 g4(B_, 2);
    lstm_rec_kernel<<<g4, 256, 0, stream>>>(Xg, whhrec, length, out);
}
